// NeighborhoodAttention2D_86096914415642
// MI455X (gfx1250) — compile-verified, matrix operand order checked by reading, not run on hardware
//
#include <hip/hip_runtime.h>
#include <hip/hip_bf16.h>

#define NH   8
#define KWIN 7
#define LR   4
#define HD   32
#define BB   4
#define HH   56
#define WW   56
#define CC   256
#define MROWS (BB*HH*WW)   /* 12544 */
#define NQKV 320           /* 32 q + 32 k + 256 v columns */

typedef __attribute__((ext_vector_type(16))) __bf16 v16bf;
typedef __attribute__((ext_vector_type(8)))  __bf16 v8bf;
typedef __attribute__((ext_vector_type(4)))  __bf16 v4bf;
typedef __attribute__((ext_vector_type(8)))  float  v8f;

// ---------------------------------------------------------------------------
// Fragment loader for pre-packed bf16 row-major data (A: M x K, B^T: N x K).
// gfx1250 16-bit A/B layout, wave32:
//   lanes 0-15  : row r, K = {k0+0..7} in v0..3 and {k0+16..23} in v4..7
//   lanes 16-31 : row r, K offsets shifted by +8
// Both K-chunks are 8 contiguous bf16 = one 16B global_load_b128 each.
// ---------------------------------------------------------------------------
__device__ __forceinline__ v16bf load_frag(const __bf16* __restrict__ base,
                                           int r, int k0, int grp) {
  const __bf16* p = base + (size_t)r * CC + k0 + grp * 8;
  v8bf lo = *(const v8bf*)(p);
  v8bf hi = *(const v8bf*)(p + 16);
  v16bf f;
#pragma unroll
  for (int i = 0; i < 8; ++i) { f[i] = lo[i]; f[8 + i] = hi[i]; }
  return f;
}

// ---------------------------------------------------------------------------
// Pack kernels (run once, tiny): weights -> transposed bf16, x -> bf16.
// Wt rows: [0,32)=Wq^T, [32,64)=Wk^T, [64,320)=Wv^T.  Wpt = Wp^T.
// ---------------------------------------------------------------------------
__global__ void __launch_bounds__(256)
pack_w_kernel(const float* __restrict__ Wq, const float* __restrict__ Wk,
              const float* __restrict__ Wv, const float* __restrict__ Wp,
              __bf16* __restrict__ Wt, __bf16* __restrict__ Wpt) {
  const int idx = blockIdx.x * 256 + threadIdx.x;   // over NQKV*CC = 81920
  const int k = idx & (CC - 1);
  const int n = idx >> 8;
  float v;
  if (n < 32)      v = Wq[k * 32 + n];
  else if (n < 64) v = Wk[k * 32 + (n - 32)];
  else             v = Wv[k * CC + (n - 64)];
  Wt[idx] = (__bf16)v;
  if (n < CC) Wpt[(size_t)n * CC + k] = (__bf16)Wp[(size_t)k * CC + n];
}

__global__ void __launch_bounds__(256)
pack_x_kernel(const float* __restrict__ x, __bf16* __restrict__ xbf) {
  const int i = (blockIdx.x * 256 + threadIdx.x) * 4;   // MROWS*CC/4 threads
  const float4 v = *(const float4*)(x + i);
  v4bf o;
  o[0] = (__bf16)v.x; o[1] = (__bf16)v.y; o[2] = (__bf16)v.z; o[3] = (__bf16)v.w;
  *(v4bf*)(xbf + i) = o;
}

// ---------------------------------------------------------------------------
// Kernel 1: fused QKV projection GEMM, M=12544, N=320, K=256, bf16 WMMA.
// Block = 160 threads = 5 waves; each wave computes a 16x64 strip (4 WMMA
// tiles, A fragment reused 4x). Grid = 784 M-tiles.
//   cols [0,32)  -> q = (acc + bq) * hd^-0.5
//   cols [32,64) -> k = acc + bk
//   cols [64,320)-> v = acc + bv
// ---------------------------------------------------------------------------
__global__ void __launch_bounds__(160)
qkv_gemm_kernel(const __bf16* __restrict__ xbf, const __bf16* __restrict__ Wt,
                const float* __restrict__ bq, const float* __restrict__ bk,
                const float* __restrict__ bv,
                float* __restrict__ qbuf, float* __restrict__ kbuf,
                float* __restrict__ vbuf) {
  const int wave = threadIdx.x >> 5;        // 0..4
  const int lane = threadIdx.x & 31;
  const int grp  = lane >> 4;
  const int nl   = lane & 15;
  const int row0 = blockIdx.x * 16;
  const int colg = wave * 64;               // wave's 64-column strip

  const int arow = row0 + nl;
  v8f acc[4] = {{}, {}, {}, {}};
#pragma unroll
  for (int k0 = 0; k0 < CC; k0 += 32) {
    __builtin_prefetch(xbf + (size_t)arow * CC + k0 + 32, 0, 1);
    const v16bf a = load_frag(xbf, arow, k0, grp);
#pragma unroll
    for (int t = 0; t < 4; ++t) {
      const v16bf b = load_frag(Wt, colg + 16 * t + nl, k0, grp);
      acc[t] = __builtin_amdgcn_wmma_f32_16x16x32_bf16(false, a, false, b,
                                                       (short)0, acc[t], false, false);
    }
  }
#pragma unroll
  for (int t = 0; t < 4; ++t) {
    const int col0 = colg + 16 * t;         // tile-uniform; tiles never straddle q/k/v
    float* out; int ldo, oc0; float sc = 1.0f; const float* bias;
    if (col0 < 32)      { out = qbuf; ldo = 32; oc0 = col0;      bias = bq;
                          sc = 0.17677669529663687f; /* 32^-0.5 */ }
    else if (col0 < 64) { out = kbuf; ldo = 32; oc0 = col0 - 32; bias = bk; }
    else                { out = vbuf; ldo = CC; oc0 = col0 - 64; bias = bv; }
    const int oc = oc0 + nl;
    const float bval = bias[oc];
#pragma unroll
    for (int v = 0; v < 8; ++v) {
      const int m = row0 + v + grp * 8;
      out[(size_t)m * ldo + oc] = (acc[t][v] + bval) * sc;
    }
  }
}

// ---------------------------------------------------------------------------
// Kernel 2: neighborhood attention core. One thread per (b, head, i, j).
// q/k/v read through the reference's raw-reshape strides:
//   q[b,n,i,j,l] = qflat[b*H*W*32 + n*H*W*4 + (i*W+j)*4 + l]  (v with 32)
// Output written pre-transposed to (B,H,W, n*32+d) directly as bf16 for the
// projection GEMM.
// ---------------------------------------------------------------------------
__global__ void __launch_bounds__(256)
natten_kernel(const float* __restrict__ qbuf, const float* __restrict__ kbuf,
              const float* __restrict__ vbuf, const float* __restrict__ rpb,
              __bf16* __restrict__ abf) {
  const int idx = blockIdx.x * blockDim.x + threadIdx.x;   // B*NH*H*W = 100352
  int j = idx % WW;  int t = idx / WW;
  int i = t % HH;    t /= HH;
  int n = t % NH;    const int b = t / NH;

  const size_t qk_base = (size_t)b * HH * WW * 32 + (size_t)n * HH * WW * 4;
  const float4 qv = *(const float4*)(qbuf + qk_base + (size_t)(i * WW + j) * 4);

  const int sh = min(max(i - 3, 0), HH - KWIN);
  const int sw = min(max(j - 3, 0), WW - KWIN);

  float s[KWIN * KWIN];
  float mx = -3.0e38f;
  const float* rb = rpb + n * 13 * 13;
#pragma unroll
  for (int ki = 0; ki < KWIN; ++ki) {
    const int kh = sh + ki;
    const float* krow = kbuf + qk_base + (size_t)(kh * WW + sw) * 4;
    const float* brow = rb + (kh - i + 6) * 13 + (sw - j + 6);
#pragma unroll
    for (int kj = 0; kj < KWIN; ++kj) {
      const float4 kv = *(const float4*)(krow + kj * 4);
      float d = qv.x * kv.x + qv.y * kv.y + qv.z * kv.z + qv.w * kv.w + brow[kj];
      s[ki * KWIN + kj] = d;
      mx = fmaxf(mx, d);
    }
  }

  float sum = 0.0f;
#pragma unroll
  for (int t2 = 0; t2 < KWIN * KWIN; ++t2) { s[t2] = __expf(s[t2] - mx); sum += s[t2]; }
  const float inv = 1.0f / sum;

  float acc[HD];
#pragma unroll
  for (int d = 0; d < HD; ++d) acc[d] = 0.0f;

  const float* vbase = vbuf + (size_t)b * HH * WW * CC + (size_t)n * HH * WW * HD;
#pragma unroll
  for (int ki = 0; ki < KWIN; ++ki) {
#pragma unroll
    for (int kj = 0; kj < KWIN; ++kj) {
      const float p = s[ki * KWIN + kj] * inv;
      const float4* vv = (const float4*)(vbase + (size_t)((sh + ki) * WW + sw + kj) * HD);
#pragma unroll
      for (int q4 = 0; q4 < HD / 4; ++q4) {
        const float4 vx = vv[q4];
        acc[4*q4+0] += p * vx.x;  acc[4*q4+1] += p * vx.y;
        acc[4*q4+2] += p * vx.z;  acc[4*q4+3] += p * vx.w;
      }
    }
  }

  __bf16* op = abf + ((size_t)(b * HH + i) * WW + j) * CC + n * HD;
#pragma unroll
  for (int g = 0; g < HD / 8; ++g) {
    v8bf o;
#pragma unroll
    for (int e = 0; e < 8; ++e) o[e] = (__bf16)acc[8 * g + e];
    *(v8bf*)(op + 8 * g) = o;   // 16B store
  }
}

// ---------------------------------------------------------------------------
// Kernel 3: output projection  out = attn_out @ Wp + bp  (M=12544, N=K=256).
// Block = 128 threads = 4 waves; each wave computes a 16x64 strip.
// ---------------------------------------------------------------------------
__global__ void __launch_bounds__(128)
proj_gemm_kernel(const __bf16* __restrict__ abf, const __bf16* __restrict__ Wpt,
                 const float* __restrict__ bp, float* __restrict__ out) {
  const int wave = threadIdx.x >> 5;        // 0..3
  const int lane = threadIdx.x & 31;
  const int grp  = lane >> 4;
  const int nl   = lane & 15;
  const int row0 = blockIdx.x * 16;
  const int colg = wave * 64;

  const int arow = row0 + nl;
  v8f acc[4] = {{}, {}, {}, {}};
#pragma unroll
  for (int k0 = 0; k0 < CC; k0 += 32) {
    __builtin_prefetch(abf + (size_t)arow * CC + k0 + 32, 0, 1);
    const v16bf a = load_frag(abf, arow, k0, grp);
#pragma unroll
    for (int t = 0; t < 4; ++t) {
      const v16bf b = load_frag(Wpt, colg + 16 * t + nl, k0, grp);
      acc[t] = __builtin_amdgcn_wmma_f32_16x16x32_bf16(false, a, false, b,
                                                       (short)0, acc[t], false, false);
    }
  }
#pragma unroll
  for (int t = 0; t < 4; ++t) {
    const int col = colg + 16 * t + nl;
    const float bval = bp[col];
#pragma unroll
    for (int v = 0; v < 8; ++v) {
      const int m = row0 + v + grp * 8;
      out[(size_t)m * CC + col] = acc[t][v] + bval;
    }
  }
}

// ---------------------------------------------------------------------------
extern "C" void kernel_launch(void* const* d_in, const int* in_sizes, int n_in,
                              void* d_out, int out_size, void* d_ws, size_t ws_size,
                              hipStream_t stream) {
  const float* x   = (const float*)d_in[0];
  const float* Wq  = (const float*)d_in[1];
  const float* bq  = (const float*)d_in[2];
  const float* Wk  = (const float*)d_in[3];
  const float* bk  = (const float*)d_in[4];
  const float* Wv  = (const float*)d_in[5];
  const float* bv  = (const float*)d_in[6];
  const float* rpb = (const float*)d_in[7];
  const float* Wp  = (const float*)d_in[8];
  const float* bp  = (const float*)d_in[9];

  // f32 scratch
  float* qbuf = (float*)d_ws;                         // MROWS*32
  float* kbuf = qbuf + (size_t)MROWS * 32;            // MROWS*32
  float* vbuf = kbuf + (size_t)MROWS * 32;            // MROWS*256
  // bf16 scratch (16B-aligned offsets by construction)
  __bf16* abf = (__bf16*)(vbuf + (size_t)MROWS * CC); // MROWS*256 bf16
  __bf16* xbf = abf + (size_t)MROWS * CC;             // MROWS*256 bf16
  __bf16* Wt  = xbf + (size_t)MROWS * CC;             // 320*256 bf16
  __bf16* Wpt = Wt + (size_t)NQKV * CC;               // 256*256 bf16

  // 0) one-shot packs: weights -> transposed bf16; x -> bf16
  pack_w_kernel<<<dim3((NQKV * CC) / 256), 256, 0, stream>>>(Wq, Wk, Wv, Wp, Wt, Wpt);
  pack_x_kernel<<<dim3((MROWS * CC) / (256 * 4)), 256, 0, stream>>>(x, xbf);

  // 1) fused QKV projection: 784 M-tiles x (5 waves * 64 cols)
  qkv_gemm_kernel<<<dim3(MROWS / 16), 160, 0, stream>>>(
      xbf, Wt, bq, bk, bv, qbuf, kbuf, vbuf);

  // 2) neighborhood attention core: one thread per query*head
  natten_kernel<<<dim3((BB * NH * HH * WW) / 256), 256, 0, stream>>>(
      qbuf, kbuf, vbuf, rpb, abf);

  // 3) output projection: 784 M-tiles x (4 waves * 64 cols)
  proj_gemm_kernel<<<dim3(MROWS / 16), 128, 0, stream>>>(
      abf, Wpt, bp, (float*)d_out);
}
